// HybridNATIsotropic_52432960749594
// MI455X (gfx1250) — matrix operand order repaired and checked by reading, not compiled, not run on hardware
//
#include <hip/hip_runtime.h>
#include <hip/hip_bf16.h>

// ---------------------------------------------------------------------------
// HybridNATIsotropic forward for MI455X (gfx1250), wave32 + WMMA f16 16x16x32
// Weights pre-packed to f16 fragment order; activations produced in f16.
// GEMM hot loop: software-pipelined 10x global_load_b128 + 4x v_wmma / K-step.
// ---------------------------------------------------------------------------

typedef __attribute__((ext_vector_type(16))) _Float16 v16h;
typedef __attribute__((ext_vector_type(8)))  _Float16 v8h;
typedef __attribute__((ext_vector_type(8)))  float    v8f;

#define BN      8
#define SS      14
#define CCH     384
#define NHEADS  12
#define KNB     7          // neighborhood size
#define HDIM    32
#define NTOK    196        // SS*SS
#define MROWS   (BN*NTOK)  // 1568
#define HIDDIM  1536
#define NCLS    1000
#define QKSCALE 0.17677669529663687f
#define EPSV    1e-5f

__device__ __forceinline__ float gelu_exact(float x) {
    return 0.5f * x * (1.0f + erff(x * 0.7071067811865476f));
}
__device__ __forceinline__ int imin(int a, int b) { return a < b ? a : b; }

#define SHUF16(lo, hi) __builtin_shufflevector(lo, hi, 0,1,2,3,4,5,6,7,8,9,10,11,12,13,14,15)
#define WMMA_F16(af, bf, cc) \
    __builtin_amdgcn_wmma_f32_16x16x32_f16(false, af, false, bf, (short)0, cc, false, false)

// ---------------------------------------------------------------------------
// Weight packing: B[K,N] f32 row-major -> f16 fragment order
//   Bp[((t*N + n)*2 + h)*16 + j] = B[t*32 + h*16 + j, n]
// so a wave's (n-lane, half) B-fragment is one contiguous 32-byte block.
// ---------------------------------------------------------------------------
__global__ void pack_w(const float* __restrict__ B, _Float16* __restrict__ Bp,
                       int Kk, int Nn)
{
    const int idx = blockIdx.x * 256 + threadIdx.x;
    const int total = (Kk >> 5) * Nn * 2;
    if (idx >= total) return;
    const int h = idx & 1;
    const int n = (idx >> 1) % Nn;
    const int t = idx / (2 * Nn);
    _Float16* dst = Bp + ((size_t)(t * Nn + n) * 2 + h) * 16;
    const float* src = B + (size_t)(t * 32 + h * 16) * Nn + n;
    #pragma unroll
    for (int j = 0; j < 16; ++j) dst[j] = (_Float16)src[(size_t)j * Nn];
}

// Same, but source is transposed: B[k,n] = W[n*K + k]  (for pe_w [C,768])
__global__ void pack_wt(const float* __restrict__ W, _Float16* __restrict__ Bp,
                        int Kk, int Nn)
{
    const int idx = blockIdx.x * 256 + threadIdx.x;
    const int total = (Kk >> 5) * Nn * 2;
    if (idx >= total) return;
    const int h = idx & 1;
    const int n = (idx >> 1) % Nn;
    const int t = idx / (2 * Nn);
    _Float16* dst = Bp + ((size_t)(t * Nn + n) * 2 + h) * 16;
    const float* src = W + (size_t)n * Kk + t * 32 + h * 16;
    #pragma unroll
    for (int j = 0; j < 16; ++j) dst[j] = (_Float16)src[j];
}

__global__ void cvt_f16(const float* __restrict__ src, _Float16* __restrict__ dst, int n)
{
    const int i = blockIdx.x * 256 + threadIdx.x;
    if (i < n) dst[i] = (_Float16)src[i];
}

// ---------------------------------------------------------------------------
// Generic GEMM:  C[M,N] = act(A[M,K] @ B[K,N] + bias) (+ residual f32)
// A: f16 row-major. B: f16 packed (pack_w layout). Wave computes 16x64.
// Software-pipelined: K-tile t+1 loads issue before tile t's WMMAs.
// K multiple of 32. act: 0=none 1=GELU. outf16: 0 -> f32 C, 1 -> f16 C.
// ---------------------------------------------------------------------------
__global__ void gemm_f16wmma(const _Float16* __restrict__ Ah,
                             const _Float16* __restrict__ Bp,
                             const float* __restrict__ bias,
                             const float* __restrict__ resid,
                             void* __restrict__ Cc,
                             int M, int Nn, int Kk, int act, int outf16)
{
    const int tilesN = (Nn + 63) >> 6;
    const int tilesM = (M  + 15) >> 4;
    const int tile = blockIdx.x * 8 + (threadIdx.x >> 5);
    if (tile >= tilesM * tilesN) return;   // wave-uniform exit
    const int tm   = tile / tilesN;
    const int tn   = tile % tilesN;
    const int lane = threadIdx.x & 31;
    const int l16  = lane & 15;
    const int half = lane >> 4;

    const int mrow = imin(tm * 16 + l16, M - 1);
    int ncol[4];
    #pragma unroll
    for (int s = 0; s < 4; ++s)
        ncol[s] = imin(tn * 64 + s * 16 + l16, Nn - 1);

    const _Float16* __restrict__ arow = Ah + (size_t)mrow * Kk + half * 16;
    const _Float16* __restrict__ bb0 = Bp + ((size_t)ncol[0] * 2 + half) * 16;
    const _Float16* __restrict__ bb1 = Bp + ((size_t)ncol[1] * 2 + half) * 16;
    const _Float16* __restrict__ bb2 = Bp + ((size_t)ncol[2] * 2 + half) * 16;
    const _Float16* __restrict__ bb3 = Bp + ((size_t)ncol[3] * 2 + half) * 16;

    v8f acc[4] = {v8f{}, v8f{}, v8f{}, v8f{}};
    const int ktiles = Kk >> 5;
    const size_t bstride = (size_t)Nn * 32;

    // pipeline prologue: tile 0 fragments
    v8h  alo = *(const v8h*)(arow);
    v8h  ahi = *(const v8h*)(arow + 8);
    v16h b0  = *(const v16h*)(bb0);
    v16h b1  = *(const v16h*)(bb1);
    v16h b2  = *(const v16h*)(bb2);
    v16h b3  = *(const v16h*)(bb3);

    for (int t = 0; t < ktiles - 1; ++t) {
        const v16h af = SHUF16(alo, ahi);
        const v16h c0 = b0, c1 = b1, c2 = b2, c3 = b3;
        // prefetch tile t+1 into fresh registers
        const size_t ao = (size_t)(t + 1) * 32;
        const size_t bo = (size_t)(t + 1) * bstride;
        alo = *(const v8h*)(arow + ao);
        ahi = *(const v8h*)(arow + ao + 8);
        b0  = *(const v16h*)(bb0 + bo);
        b1  = *(const v16h*)(bb1 + bo);
        b2  = *(const v16h*)(bb2 + bo);
        b3  = *(const v16h*)(bb3 + bo);
        acc[0] = WMMA_F16(af, c0, acc[0]);
        acc[1] = WMMA_F16(af, c1, acc[1]);
        acc[2] = WMMA_F16(af, c2, acc[2]);
        acc[3] = WMMA_F16(af, c3, acc[3]);
    }
    { // pipeline epilogue: last tile
        const v16h af = SHUF16(alo, ahi);
        acc[0] = WMMA_F16(af, b0, acc[0]);
        acc[1] = WMMA_F16(af, b1, acc[1]);
        acc[2] = WMMA_F16(af, b2, acc[2]);
        acc[3] = WMMA_F16(af, b3, acc[3]);
    }

    #pragma unroll
    for (int s = 0; s < 4; ++s) {
        const int n = tn * 64 + s * 16 + l16;
        if (n < Nn) {
            const float bs = bias[n];
            #pragma unroll
            for (int r = 0; r < 8; ++r) {
                const int m = tm * 16 + half * 8 + r;   // D: vgpr r -> M=half*8+r
                if (m < M) {
                    float val = acc[s][r] + bs;
                    if (act == 1) val = gelu_exact(val);
                    if (resid)    val += resid[(size_t)m * Nn + n];
                    if (outf16) ((_Float16*)Cc)[(size_t)m * Nn + n] = (_Float16)val;
                    else        ((float*)Cc)[(size_t)m * Nn + n] = val;
                }
            }
        }
    }
}

// ---------------------------------------------------------------------------
// Patch embed GEMM: A = im2col of xh (f16), B = packed pe_w, out f32 + bias.
// All dims exact multiples -> guard-free. M=1568, N=384, K=768.
// A-fragment = two contiguous 8-half runs. Software-pipelined like above.
// ---------------------------------------------------------------------------
__device__ __forceinline__ const _Float16* pe_arun(const _Float16* pixbase, int kg) {
    const int ci = kg >> 8, ph = (kg >> 4) & 15, pw = kg & 15;
    return pixbase + ((size_t)ci * 224 + ph) * 224 + pw;
}

__global__ void gemm_patch_embed(const _Float16* __restrict__ xh,
                                 const _Float16* __restrict__ wp,
                                 const float* __restrict__ bias,
                                 float* __restrict__ out)
{
    const int tilesN = CCH / 64;                  // 6
    const int ntiles = (MROWS / 16) * tilesN;     // 98*6
    const int tile = blockIdx.x * 8 + (threadIdx.x >> 5);
    if (tile >= ntiles) return;
    const int tm   = tile / tilesN;
    const int tn   = tile % tilesN;
    const int lane = threadIdx.x & 31;
    const int l16  = lane & 15;
    const int half = lane >> 4;
    const int mrow = tm * 16 + l16;
    const int bimg = mrow / NTOK;
    const int p    = mrow % NTOK;
    const int py   = p / SS, px = p % SS;
    const _Float16* __restrict__ pixbase =
        xh + (size_t)bimg * 3 * 224 * 224 + (size_t)(py * 16) * 224 + px * 16;

    const _Float16* __restrict__ bb0 = wp + ((size_t)(tn * 64 +  0 + l16) * 2 + half) * 16;
    const _Float16* __restrict__ bb1 = wp + ((size_t)(tn * 64 + 16 + l16) * 2 + half) * 16;
    const _Float16* __restrict__ bb2 = wp + ((size_t)(tn * 64 + 32 + l16) * 2 + half) * 16;
    const _Float16* __restrict__ bb3 = wp + ((size_t)(tn * 64 + 48 + l16) * 2 + half) * 16;
    const size_t bstride = (size_t)CCH * 32;
    const int ktiles = 24;                        // 768/32

    v8f acc[4] = {v8f{}, v8f{}, v8f{}, v8f{}};

    v8h  alo = *(const v8h*)pe_arun(pixbase, half * 16);
    v8h  ahi = *(const v8h*)pe_arun(pixbase, half * 16 + 8);
    v16h b0  = *(const v16h*)(bb0);
    v16h b1  = *(const v16h*)(bb1);
    v16h b2  = *(const v16h*)(bb2);
    v16h b3  = *(const v16h*)(bb3);

    for (int t = 0; t < ktiles - 1; ++t) {
        const v16h af = SHUF16(alo, ahi);
        const v16h c0 = b0, c1 = b1, c2 = b2, c3 = b3;
        const int kg = (t + 1) * 32 + half * 16;
        const size_t bo = (size_t)(t + 1) * bstride;
        alo = *(const v8h*)pe_arun(pixbase, kg);
        ahi = *(const v8h*)pe_arun(pixbase, kg + 8);
        b0  = *(const v16h*)(bb0 + bo);
        b1  = *(const v16h*)(bb1 + bo);
        b2  = *(const v16h*)(bb2 + bo);
        b3  = *(const v16h*)(bb3 + bo);
        acc[0] = WMMA_F16(af, c0, acc[0]);
        acc[1] = WMMA_F16(af, c1, acc[1]);
        acc[2] = WMMA_F16(af, c2, acc[2]);
        acc[3] = WMMA_F16(af, c3, acc[3]);
    }
    {
        const v16h af = SHUF16(alo, ahi);
        acc[0] = WMMA_F16(af, b0, acc[0]);
        acc[1] = WMMA_F16(af, b1, acc[1]);
        acc[2] = WMMA_F16(af, b2, acc[2]);
        acc[3] = WMMA_F16(af, b3, acc[3]);
    }

    #pragma unroll
    for (int s = 0; s < 4; ++s) {
        const int n = tn * 64 + s * 16 + l16;
        const float bs = bias[n];
        #pragma unroll
        for (int r = 0; r < 8; ++r) {
            const int m = tm * 16 + half * 8 + r;
            out[(size_t)m * CCH + n] = acc[s][r] + bs;
        }
    }
}

// ---------------------------------------------------------------------------
// LayerNorm over last dim (384). One wave32 per row, 8 rows per block.
// outf16: 1 -> f16 output (GEMM input), 0 -> f32 (residual stream)
// ---------------------------------------------------------------------------
__global__ void ln_rows(const float* __restrict__ x, const float* __restrict__ g,
                        const float* __restrict__ b, void* __restrict__ out,
                        int rows, int outf16)
{
    const int wave = threadIdx.x >> 5;
    const int lane = threadIdx.x & 31;
    const int row  = blockIdx.x * 8 + wave;
    if (row >= rows) return;
    const float* xr = x + (size_t)row * CCH;
    float vals[12];
    float s = 0.f, sq = 0.f;
    #pragma unroll
    for (int i = 0; i < 12; ++i) {
        float v = xr[lane + i * 32];
        vals[i] = v; s += v; sq += v * v;
    }
    #pragma unroll
    for (int off = 16; off > 0; off >>= 1) {
        s  += __shfl_xor(s,  off, 32);
        sq += __shfl_xor(sq, off, 32);
    }
    const float mean = s * (1.f / CCH);
    const float var  = sq * (1.f / CCH) - mean * mean;
    const float rstd = rsqrtf(var + EPSV);
    _Float16* o16 = (_Float16*)out + (size_t)row * CCH;
    float*    o32 = (float*)out    + (size_t)row * CCH;
    #pragma unroll
    for (int i = 0; i < 12; ++i) {
        const int c = lane + i * 32;
        const float val = (vals[i] - mean) * rstd * g[c] + b[c];
        if (outf16) o16[c] = (_Float16)val;
        else        o32[c] = val;
    }
}

// ---------------------------------------------------------------------------
// Neighborhood attention start/pad-base index (mirrors _na_idx)
// ---------------------------------------------------------------------------
__device__ __forceinline__ void na_sp(int i, int d, int& s, int& p) {
    const int nh = KNB / 2;
    if (d <= 1) {
        s = i - nh; if (s < 0) s = 0;
        if (i + nh >= SS) s += SS - i - nh - 1;
        p = nh;
        if (i < nh) p += nh - i;
        if (i + nh >= SS) p += SS - i - 1 - nh;
    } else if (i - nh * d < 0) {
        s = i % d;
        p = KNB - 1 - i / d;
    } else if (i + nh * d >= SS) {
        const int im = i % d;
        const int a  = (SS / d) * d;
        const int bb = SS - a;
        s = (im < bb) ? (SS - bb + im - 2 * nh * d) : (a + im - KNB * d);
        p = (SS - i - 1) / d;
    } else {
        s = i - nh * d;
        p = nh;
    }
}

// ---------------------------------------------------------------------------
// Local neighborhood attention: one thread per (batch, head, query pixel).
// qkv f32 [m][t*384 + h*32 + d]; output f16 (next GEMM's A).
// ---------------------------------------------------------------------------
__global__ void na_attn(const float* __restrict__ qkv, const float* __restrict__ rpb,
                        _Float16* __restrict__ o, int dil)
{
    const int tid = blockIdx.x * blockDim.x + threadIdx.x;
    if (tid >= BN * NHEADS * NTOK) return;
    const int p  = tid % NTOK;
    const int hh = (tid / NTOK) % NHEADS;
    const int bb = tid / (NTOK * NHEADS);
    const int i = p / SS, j = p % SS;
    int si, pi, sj, pj;
    na_sp(i, dil, si, pi);
    na_sp(j, dil, sj, pj);
    const int m0 = bb * NTOK + p;
    const float* qrow = qkv + (size_t)m0 * (3 * CCH) + hh * HDIM;
    float q[HDIM];
    #pragma unroll
    for (int d = 0; d < HDIM; ++d) q[d] = qrow[d] * QKSCALE;

    float logits[KNB * KNB];
    float mx = -1e30f;
    #pragma unroll
    for (int a = 0; a < KNB; ++a) {
        const int ki = si + a * dil;
        #pragma unroll
        for (int c = 0; c < KNB; ++c) {
            const int kj = sj + c * dil;
            const float* krow = qkv + (size_t)(bb * NTOK + ki * SS + kj) * (3 * CCH)
                                + CCH + hh * HDIM;
            float dot = 0.f;
            #pragma unroll
            for (int d = 0; d < HDIM; ++d) dot += q[d] * krow[d];
            dot += rpb[(hh * 13 + (pi + a)) * 13 + (pj + c)];
            logits[a * KNB + c] = dot;
            mx = fmaxf(mx, dot);
        }
    }
    float se = 0.f;
    #pragma unroll
    for (int t = 0; t < KNB * KNB; ++t) { logits[t] = __expf(logits[t] - mx); se += logits[t]; }
    const float inv = 1.f / se;

    float acc[HDIM];
    #pragma unroll
    for (int d = 0; d < HDIM; ++d) acc[d] = 0.f;
    #pragma unroll
    for (int a = 0; a < KNB; ++a) {
        const int ki = si + a * dil;
        #pragma unroll
        for (int c = 0; c < KNB; ++c) {
            const int kj = sj + c * dil;
            const float w = logits[a * KNB + c] * inv;
            const float* vrow = qkv + (size_t)(bb * NTOK + ki * SS + kj) * (3 * CCH)
                                + 2 * CCH + hh * HDIM;
            #pragma unroll
            for (int d = 0; d < HDIM; ++d) acc[d] += w * vrow[d];
        }
    }
    _Float16* orow = o + (size_t)m0 * CCH + hh * HDIM;
    #pragma unroll
    for (int d = 0; d < HDIM; ++d) orow[d] = (_Float16)acc[d];
}

// ---------------------------------------------------------------------------
// Global attention over 196 tokens, flash-style online softmax, f16 out.
// bias[h,q,k] = g_rpb[h, ky-qy+13, kx-qx+13]
// ---------------------------------------------------------------------------
__global__ void glob_attn(const float* __restrict__ qkv, const float* __restrict__ rpb,
                          _Float16* __restrict__ o)
{
    const int tid = blockIdx.x * blockDim.x + threadIdx.x;
    if (tid >= BN * NHEADS * NTOK) return;
    const int qp = tid % NTOK;
    const int hh = (tid / NTOK) % NHEADS;
    const int bb = tid / (NTOK * NHEADS);
    const int qy = qp / SS, qx = qp % SS;
    const int m0 = bb * NTOK + qp;
    const float* qrow = qkv + (size_t)m0 * (3 * CCH) + hh * HDIM;
    float q[HDIM];
    #pragma unroll
    for (int d = 0; d < HDIM; ++d) q[d] = qrow[d] * QKSCALE;

    float mx = -1e30f, l = 0.f;
    float acc[HDIM];
    #pragma unroll
    for (int d = 0; d < HDIM; ++d) acc[d] = 0.f;
    const float* base = qkv + (size_t)bb * NTOK * (3 * CCH);
    for (int kp = 0; kp < NTOK; ++kp) {
        const float* krow = base + (size_t)kp * (3 * CCH) + CCH + hh * HDIM;
        float dot = 0.f;
        #pragma unroll
        for (int d = 0; d < HDIM; ++d) dot += q[d] * krow[d];
        const int ky = kp / SS, kx = kp % SS;
        dot += rpb[(hh * 27 + (ky - qy + 13)) * 27 + (kx - qx + 13)];
        const float nm   = fmaxf(mx, dot);
        const float corr = __expf(mx - nm);
        const float w    = __expf(dot - nm);
        l = l * corr + w;
        const float* vrow = base + (size_t)kp * (3 * CCH) + 2 * CCH + hh * HDIM;
        #pragma unroll
        for (int d = 0; d < HDIM; ++d) acc[d] = acc[d] * corr + w * vrow[d];
        mx = nm;
    }
    const float inv = 1.f / l;
    _Float16* orow = o + (size_t)m0 * CCH + hh * HDIM;
    #pragma unroll
    for (int d = 0; d < HDIM; ++d) orow[d] = (_Float16)(acc[d] * inv);
}

// ---------------------------------------------------------------------------
// Mean pool over tokens: pooled[b, c] = mean_p h[(b*196+p), c]  (f16 -> f16)
// ---------------------------------------------------------------------------
__global__ void mean_pool(const _Float16* __restrict__ h, _Float16* __restrict__ pooled)
{
    const int c  = threadIdx.x;   // 384
    const int bb = blockIdx.x;    // 8
    float s = 0.f;
    for (int r = 0; r < NTOK; ++r) s += (float)h[((size_t)bb * NTOK + r) * CCH + c];
    pooled[bb * CCH + c] = (_Float16)(s * (1.f / NTOK));
}

// ---------------------------------------------------------------------------
extern "C" void kernel_launch(void* const* d_in, const int* in_sizes, int n_in,
                              void* d_out, int out_size, void* d_ws, size_t ws_size,
                              hipStream_t stream) {
    (void)in_sizes; (void)n_in; (void)out_size; (void)ws_size;
    const float* x     = (const float*)d_in[0];
    const float* pe_w  = (const float*)d_in[1];
    const float* pe_b  = (const float*)d_in[2];
    const float* pe_ng = (const float*)d_in[3];
    const float* pe_nb = (const float*)d_in[4];
    const float* l_n1g = (const float*)d_in[5];
    const float* l_n1b = (const float*)d_in[6];
    const float* l_qw  = (const float*)d_in[7];
    const float* l_qb  = (const float*)d_in[8];
    const float* l_rpb = (const float*)d_in[9];
    const float* l_pw  = (const float*)d_in[10];
    const float* l_pb  = (const float*)d_in[11];
    const float* l_n2g = (const float*)d_in[12];
    const float* l_n2b = (const float*)d_in[13];
    const float* l_f1w = (const float*)d_in[14];
    const float* l_f1b = (const float*)d_in[15];
    const float* l_f2w = (const float*)d_in[16];
    const float* l_f2b = (const float*)d_in[17];
    const float* g_n1g = (const float*)d_in[18];
    const float* g_n1b = (const float*)d_in[19];
    const float* g_qw  = (const float*)d_in[20];
    const float* g_qb  = (const float*)d_in[21];
    const float* g_rpb = (const float*)d_in[22];
    const float* g_pw  = (const float*)d_in[23];
    const float* g_pb  = (const float*)d_in[24];
    const float* g_n2g = (const float*)d_in[25];
    const float* g_n2b = (const float*)d_in[26];
    const float* g_f1w = (const float*)d_in[27];
    const float* g_f1b = (const float*)d_in[28];
    const float* g_f2w = (const float*)d_in[29];
    const float* g_f2b = (const float*)d_in[30];
    const float* fn_g  = (const float*)d_in[31];
    const float* fn_b  = (const float*)d_in[32];
    const float* h_w   = (const float*)d_in[33];
    const float* h_b   = (const float*)d_in[34];
    float* out = (float*)d_out;

    // ---- workspace layout ----
    float* y   = (float*)d_ws;                         // f32 1568*384
    float* qkv = y + (size_t)MROWS * CCH;              // f32 1568*1152 (also pe temp)
    _Float16* hp = (_Float16*)(qkv + (size_t)MROWS * 3 * CCH);
    _Float16* hbuf   = hp;  hp += (size_t)MROWS * CCH;
    _Float16* obuf   = hp;  hp += (size_t)MROWS * CCH;
    _Float16* hid    = hp;  hp += (size_t)MROWS * HIDDIM;
    _Float16* pooled = hp;  hp += BN * CCH;
    _Float16* xh     = hp;  hp += (size_t)BN * 3 * 224 * 224;
    _Float16* pe_wp  = hp;  hp += (size_t)768 * CCH;
    _Float16* l_qwp  = hp;  hp += (size_t)8 * CCH * 3 * CCH;
    _Float16* l_pwp  = hp;  hp += (size_t)8 * CCH * CCH;
    _Float16* l_f1wp = hp;  hp += (size_t)8 * CCH * HIDDIM;
    _Float16* l_f2wp = hp;  hp += (size_t)8 * HIDDIM * CCH;
    _Float16* g_qwp  = hp;  hp += (size_t)4 * CCH * 3 * CCH;
    _Float16* g_pwp  = hp;  hp += (size_t)4 * CCH * CCH;
    _Float16* g_f1wp = hp;  hp += (size_t)4 * CCH * HIDDIM;
    _Float16* g_f2wp = hp;  hp += (size_t)4 * HIDDIM * CCH;
    _Float16* h_wp   = hp;  hp += (size_t)(CCH / 32) * NCLS * 32;

    auto pack = [&](const float* src, _Float16* dst, int Kk, int Nn) {
        const int total = (Kk / 32) * Nn * 2;
        pack_w<<<(total + 255) / 256, 256, 0, stream>>>(src, dst, Kk, Nn);
    };

    // ---- one-time (per call) weight packing / input conversion ----
    {
        const int nx = BN * 3 * 224 * 224;
        cvt_f16<<<(nx + 255) / 256, 256, 0, stream>>>(x, xh, nx);
        const int tpe = (768 / 32) * CCH * 2;
        pack_wt<<<(tpe + 255) / 256, 256, 0, stream>>>(pe_w, pe_wp, 768, CCH);
        for (int i = 0; i < 8; ++i) {
            pack(l_qw  + (size_t)i * CCH * 3 * CCH, l_qwp  + (size_t)i * CCH * 3 * CCH, CCH, 3 * CCH);
            pack(l_pw  + (size_t)i * CCH * CCH,     l_pwp  + (size_t)i * CCH * CCH,     CCH, CCH);
            pack(l_f1w + (size_t)i * CCH * HIDDIM,  l_f1wp + (size_t)i * CCH * HIDDIM,  CCH, HIDDIM);
            pack(l_f2w + (size_t)i * HIDDIM * CCH,  l_f2wp + (size_t)i * HIDDIM * CCH,  HIDDIM, CCH);
        }
        for (int i = 0; i < 4; ++i) {
            pack(g_qw  + (size_t)i * CCH * 3 * CCH, g_qwp  + (size_t)i * CCH * 3 * CCH, CCH, 3 * CCH);
            pack(g_pw  + (size_t)i * CCH * CCH,     g_pwp  + (size_t)i * CCH * CCH,     CCH, CCH);
            pack(g_f1w + (size_t)i * CCH * HIDDIM,  g_f1wp + (size_t)i * CCH * HIDDIM,  CCH, HIDDIM);
            pack(g_f2w + (size_t)i * HIDDIM * CCH,  g_f2wp + (size_t)i * HIDDIM * CCH,  HIDDIM, CCH);
        }
        pack(h_w, h_wp, CCH, NCLS);
    }

    auto gemm = [&](const _Float16* A, const _Float16* Bp, const float* bias,
                    const float* resid, void* Cd, int M, int Nn, int Kk,
                    int act, int outf16) {
        const int tiles  = ((M + 15) / 16) * ((Nn + 63) / 64);
        const int blocks = (tiles + 7) / 8;
        gemm_f16wmma<<<blocks, 256, 0, stream>>>(A, Bp, bias, resid, Cd,
                                                 M, Nn, Kk, act, outf16);
    };

    const int attn_threads = BN * NHEADS * NTOK;           // 18816
    const int attn_blocks  = (attn_threads + 255) / 256;

    // ---- patch embed: conv-as-GEMM + bias -> qkv(temp f32), LN -> y (f32) ----
    {
        const int tiles  = (MROWS / 16) * (CCH / 64);
        const int blocks = (tiles + 7) / 8;
        gemm_patch_embed<<<blocks, 256, 0, stream>>>(xh, pe_wp, pe_b, qkv);
    }
    ln_rows<<<196, 256, 0, stream>>>(qkv, pe_ng, pe_nb, y, MROWS, 0);

    // ---- 8 local NA layers ----
    for (int i = 0; i < 8; ++i) {
        const int dil = (i % 2 == 0) ? 1 : 2;
        ln_rows<<<196, 256, 0, stream>>>(y, l_n1g + i * CCH, l_n1b + i * CCH, hbuf, MROWS, 1);
        gemm(hbuf, l_qwp + (size_t)i * CCH * 3 * CCH, l_qb + (size_t)i * 3 * CCH,
             nullptr, qkv, MROWS, 3 * CCH, CCH, 0, 0);
        na_attn<<<attn_blocks, 256, 0, stream>>>(qkv, l_rpb + (size_t)i * NHEADS * 13 * 13,
                                                 obuf, dil);
        gemm(obuf, l_pwp + (size_t)i * CCH * CCH, l_pb + (size_t)i * CCH,
             y, y, MROWS, CCH, CCH, 0, 0);
        ln_rows<<<196, 256, 0, stream>>>(y, l_n2g + i * CCH, l_n2b + i * CCH, hbuf, MROWS, 1);
        gemm(hbuf, l_f1wp + (size_t)i * CCH * HIDDIM, l_f1b + (size_t)i * HIDDIM,
             nullptr, hid, MROWS, HIDDIM, CCH, 1, 1);
        gemm(hid, l_f2wp + (size_t)i * HIDDIM * CCH, l_f2b + (size_t)i * CCH,
             y, y, MROWS, CCH, HIDDIM, 0, 0);
    }

    // ---- 4 global attention layers ----
    for (int i = 0; i < 4; ++i) {
        ln_rows<<<196, 256, 0, stream>>>(y, g_n1g + i * CCH, g_n1b + i * CCH, hbuf, MROWS, 1);
        gemm(hbuf, g_qwp + (size_t)i * CCH * 3 * CCH, g_qb + (size_t)i * 3 * CCH,
             nullptr, qkv, MROWS, 3 * CCH, CCH, 0, 0);
        glob_attn<<<attn_blocks, 256, 0, stream>>>(qkv, g_rpb + (size_t)i * NHEADS * 27 * 27,
                                                   obuf);
        gemm(obuf, g_pwp + (size_t)i * CCH * CCH, g_pb + (size_t)i * CCH,
             y, y, MROWS, CCH, CCH, 0, 0);
        ln_rows<<<196, 256, 0, stream>>>(y, g_n2g + i * CCH, g_n2b + i * CCH, hbuf, MROWS, 1);
        gemm(hbuf, g_f1wp + (size_t)i * CCH * HIDDIM, g_f1b + (size_t)i * HIDDIM,
             nullptr, hid, MROWS, HIDDIM, CCH, 1, 1);
        gemm(hid, g_f2wp + (size_t)i * HIDDIM * CCH, g_f2b + (size_t)i * CCH,
             y, y, MROWS, CCH, HIDDIM, 0, 0);
    }

    // ---- final LN -> mean pool -> classifier head ----
    ln_rows<<<196, 256, 0, stream>>>(y, fn_g, fn_b, hbuf, MROWS, 1);
    mean_pool<<<BN, CCH, 0, stream>>>(hbuf, pooled);
    gemm(pooled, h_wp, h_b, nullptr, out, BN, NCLS, CCH, 0, 0);
}
